// SuperFrame_1400159338932
// MI455X (gfx1250) — compile-verified, matrix operand order checked by reading, not run on hardware
//
#include <hip/hip_runtime.h>
#include <hip/hip_bf16.h>

// ---- WMMA vector types (probe-confirmed shapes for gfx1250) ----
typedef __attribute__((ext_vector_type(16))) __bf16         v16bf;
typedef __attribute__((ext_vector_type(8)))  float          v8f;
typedef __attribute__((ext_vector_type(8)))  unsigned short us8;
typedef __attribute__((ext_vector_type(4)))  unsigned short us4;

// ---- Problem constants (B=32, T=8192, F=80, ODIM=1024, IDIM=640) ----
#define TOUT        2728                     // output time steps
#define TT          2                        // t-tiles per block
#define NBLK        (TOUT / TT)              // 1364 blocks
#define IDIM        640
#define ODIM        1024
#define SPAN        (IDIM + 240 * (TT - 1))  // 880 contiguous floats cover 2 t-tiles
#define ASTRIDE     (SPAN + 8)               // 888 halves: stride mod 64 banks = 60
#define SRC_ROW     (8192 * 80)              // 655360 floats per batch row
#define OUT_ELEMS   ((size_t)TOUT * 16 * ODIM)  // 44,695,552 floats of x_out

static __device__ __forceinline__ unsigned short f2bf(float x) {
  unsigned int u = __float_as_uint(x);
  u += 0x7FFFu + ((u >> 16) & 1u);
  return (unsigned short)(u >> 16);
}

static __device__ __forceinline__ v16bf load_frag(const unsigned short* p) {
  // Two contiguous 16B chunks: K {0..7} and K {+16..+23} relative to lane base.
  union { us8 u[2]; v16bf v; } f;
  f.u[0] = *(const us8*)(p);
  f.u[1] = *(const us8*)(p + 16);
  return f.v;
}

// ---------------------------------------------------------------------------
// Prep: W (1024x640 f32, row-major) -> bf16 in workspace; emit out_lengths.
// ---------------------------------------------------------------------------
__global__ void prep_kernel(const float* __restrict__ W,
                            const long long* __restrict__ lens,
                            unsigned short* __restrict__ Wbf,
                            float* __restrict__ outLens) {
  const int i = (blockIdx.x * 256 + threadIdx.x) * 4;
  if (i < ODIM * IDIM) {
    const float4 w = *(const float4*)(W + i);
    us4 o;
    o.x = f2bf(w.x); o.y = f2bf(w.y); o.z = f2bf(w.z); o.w = f2bf(w.w);
    *(us4*)(Wbf + i) = o;
  }
  if (blockIdx.x == 0 && threadIdx.x < 32) {
    outLens[threadIdx.x] = (float)(lens[threadIdx.x] / 3 - 2);
  }
}

// ---------------------------------------------------------------------------
// Main: per block, 2 t-tiles. A(32 x 880 span) staged to LDS as bf16 (56.8 KB).
// 8 waves; each wave: 2 groups of 4 N-tiles. Per kk-step: 4 A-fragments and
// 4 B-fragments -> 16 WMMAs (each A reused 4x across nt, each B 4x across
// dt/mr). acc = 16 x v8f = 128 VGPRs. Opaque LDS offset blocks A-fragment
// hoisting across groups (round-2 spill pathology). GLU fused in epilogue.
// ---------------------------------------------------------------------------
__global__ void __launch_bounds__(256, 2)
superframe_glu_gemm(const float* __restrict__ src,
                    const unsigned short* __restrict__ Wbf,
                    const float* __restrict__ bias,
                    float* __restrict__ out) {
  __shared__ unsigned short smem[32 * ASTRIDE];   // 56,832 B

  const int tid      = threadIdx.x;
  const int t_base   = blockIdx.x * TT;
  const int src_base = t_base * 240;              // float offset of window start

  // ---- Stage A span to LDS (f32 -> bf16), float4 granularity ----
  for (int idx = tid; idx < 32 * (SPAN / 4); idx += 256) {
    const int row = idx / (SPAN / 4);
    const int kp  = (idx - row * (SPAN / 4)) * 4;
    const float4 v = *(const float4*)(src + (size_t)row * SRC_ROW + src_base + kp);
    us4 o;
    o.x = f2bf(v.x); o.y = f2bf(v.y); o.z = f2bf(v.z); o.w = f2bf(v.w);
    *(us4*)(&smem[row * ASTRIDE + kp]) = o;
  }
  __syncthreads();

  const int lane  = tid & 31;
  const int wave  = tid >> 5;
  const int lhalf = lane & 15;       // M (A) / N (B/C/D) index within tile
  const int khalf = lane >> 4;       // which K half-group this lane holds

  for (int jg = 0; jg < 2; ++jg) {   // 2 groups of 4 N-tiles per wave
    const int n0 = wave * 128 + jg * 64;

    // Opaque zero: stops the compiler from hoisting A LDS loads across groups.
    int smoff = 0;
    asm volatile("" : "+v"(smoff));

    float bv[4];
    const unsigned short* wb[4];
#pragma unroll
    for (int nt = 0; nt < 4; ++nt) {
      bv[nt] = bias[n0 + nt * 16 + lhalf];
      wb[nt] = Wbf + (size_t)(n0 + nt * 16 + lhalf) * IDIM + khalf * 8;
    }

    v8f acc[4][TT][2];
#pragma unroll
    for (int nt = 0; nt < 4; ++nt)
#pragma unroll
      for (int dt = 0; dt < TT; ++dt) {
        acc[nt][dt][0] = (v8f){0.f, 0.f, 0.f, 0.f, 0.f, 0.f, 0.f, 0.f};
        acc[nt][dt][1] = (v8f){0.f, 0.f, 0.f, 0.f, 0.f, 0.f, 0.f, 0.f};
      }

#pragma unroll 1
    for (int kk = 0; kk < IDIM / 32; ++kk) {        // 20 K-steps of 32
      v16bf b[4];
#pragma unroll
      for (int nt = 0; nt < 4; ++nt) b[nt] = load_frag(wb[nt] + kk * 32);
#pragma unroll
      for (int dt = 0; dt < TT; ++dt) {
#pragma unroll
        for (int mr = 0; mr < 2; ++mr) {            // 0: linear rows, 1: gate rows
          const v16bf a = load_frag(
              &smem[smoff + (mr * 16 + lhalf) * ASTRIDE + dt * 240 + kk * 32 + khalf * 8]);
#pragma unroll
          for (int nt = 0; nt < 4; ++nt) {
            acc[nt][dt][mr] = __builtin_amdgcn_wmma_f32_16x16x32_bf16(
                false, a, false, b[nt], (short)0, acc[nt][dt][mr], false, false);
          }
        }
      }
    }

    // ---- Fused GLU epilogue: out[t, m, n] = (lin+b) * sigmoid(gate+b) ----
#pragma unroll
    for (int nt = 0; nt < 4; ++nt) {
#pragma unroll
      for (int dt = 0; dt < TT; ++dt) {
        float* op = out + ((size_t)(t_base + dt) * 16) * ODIM + (n0 + nt * 16) + lhalf;
#pragma unroll
        for (int r = 0; r < 8; ++r) {
          const int   M    = r + khalf * 8;         // C/D layout: M = vgpr + 8*half
          const float lin  = acc[nt][dt][0][r] + bv[nt];
          const float gate = acc[nt][dt][1][r] + bv[nt];
          op[(size_t)M * ODIM] = lin * (1.0f / (1.0f + __expf(-gate)));
        }
      }
    }
  }
}

// ---------------------------------------------------------------------------
extern "C" void kernel_launch(void* const* d_in, const int* in_sizes, int n_in,
                              void* d_out, int out_size, void* d_ws, size_t ws_size,
                              hipStream_t stream) {
  const float*     src  = (const float*)d_in[0];      // (32, 8192, 80) f32
  const long long* lens = (const long long*)d_in[1];  // (32,) i64
  const float*     W    = (const float*)d_in[2];      // (1024, 640) f32
  const float*     bias = (const float*)d_in[3];      // (1024,) f32
  float*           out  = (float*)d_out;              // x_out (2728,16,1024) ++ out_lengths(32)
  unsigned short*  Wbf  = (unsigned short*)d_ws;      // 1.31 MB bf16 W

  prep_kernel<<<(ODIM * IDIM / 4 + 255) / 256, 256, 0, stream>>>(
      W, lens, Wbf, out + OUT_ELEMS);

  superframe_glu_gemm<<<NBLK, 256, 0, stream>>>(src, Wbf, bias, out);
}
